// ProgressiveInteractionLayer_33011118637072
// MI455X (gfx1250) — compile-verified, matrix operand order checked by reading, not run on hardware
//
#include <hip/hip_runtime.h>

// ---------------------------------------------------------------------------
// ProgressiveInteractionLayer for MI455X (gfx1250, wave32, WMMA)
//
// softmax over a length-1 axis == identity, so Q/K are dead. Per side:
//   att  = (kv @ Wv^T + bv) @ Wout^T + bout
//   out  = LayerNorm(q + scale * att)
// bf16 WMMA GEMMs (f32 accum) with double-buffered ASYNC global->LDS copies
// (global_load_async_to_lds_b128 + s_wait_asynccnt), fused epilogues, row LN.
// ---------------------------------------------------------------------------

typedef __attribute__((ext_vector_type(16))) __bf16        v16bf;
typedef __attribute__((ext_vector_type(8)))  float         v8f;
typedef __attribute__((ext_vector_type(8)))  unsigned int  v8u;

#define E_DIM 1024
#define B_DIM 32768
#define LN_EPS 1e-5f

#define TM 128
#define TN 128
#define TKL 64                     // LDS k-tile (2 WMMA k-steps)
#define NKT (E_DIM / TKL)          // 16 k-tiles

__device__ __forceinline__ unsigned short f32_to_bf16(float f) {
  unsigned int u = __builtin_bit_cast(unsigned int, f);
  u += 0x7FFFu + ((u >> 16) & 1u);               // round-to-nearest-even
  return (unsigned short)(u >> 16);
}

// 64B per-lane async copy: global -> LDS, tracked by ASYNCcnt.
// INST_OFFSET is added to both the LDS and the global address (ISA 10.7).
#define ASYNC_COPY64B(ldsOff, gptr)                                            \
  asm volatile("global_load_async_to_lds_b128 %0, %1, off\n\t"                 \
               "global_load_async_to_lds_b128 %0, %1, off offset:16\n\t"       \
               "global_load_async_to_lds_b128 %0, %1, off offset:32\n\t"       \
               "global_load_async_to_lds_b128 %0, %1, off offset:48"           \
               :: "v"(ldsOff), "v"(gptr) : "memory")

// ------------------------- f32 -> bf16 packing ------------------------------
__global__ __launch_bounds__(256) void pack_bf16_kernel(
    const float* __restrict__ src, unsigned short* __restrict__ dst, int n8) {
  int i = blockIdx.x * blockDim.x + threadIdx.x;
  if (i >= n8) return;
  const float4* s = reinterpret_cast<const float4*>(src) + (size_t)i * 2;
  float4 a = s[0];
  float4 b = s[1];
  uint4 o;
  o.x = (unsigned)f32_to_bf16(a.x) | ((unsigned)f32_to_bf16(a.y) << 16);
  o.y = (unsigned)f32_to_bf16(a.z) | ((unsigned)f32_to_bf16(a.w) << 16);
  o.z = (unsigned)f32_to_bf16(b.x) | ((unsigned)f32_to_bf16(b.y) << 16);
  o.w = (unsigned)f32_to_bf16(b.z) | ((unsigned)f32_to_bf16(b.w) << 16);
  reinterpret_cast<uint4*>(dst)[i] = o;
}

// ----------------------- shared GEMM mainloop body --------------------------
// 128x128 tile of C = A(MxK) @ W(NxK)^T, bf16 WMMA, f32 accum.
// 8 waves/block; wave (wm,wn) owns 64x32 = 4x2 WMMA tiles.
// Double-buffered LDS, filled by async global->LDS DMA; tile t+1 streams in
// while tile t feeds the WMMA pipe.
#define GEMM_MAINLOOP(A_, W_)                                                  \
  __shared__ __align__(32) unsigned short sA[2][TM * TKL];                     \
  __shared__ __align__(32) unsigned short sB[2][TN * TKL];                     \
  const int tid  = threadIdx.x;                                               \
  const int lane = tid & 31;                                                  \
  const int wv   = tid >> 5;                                                  \
  const int wm   = wv >> 2;      /* 0..1 : 64-row slab */                     \
  const int wn   = wv & 3;       /* 0..3 : 32-col slab */                     \
  const int blockM = blockIdx.y * TM;                                         \
  const int blockN = blockIdx.x * TN;                                         \
  const int ldRow = tid >> 1;            /* 0..127             */             \
  const int ldK   = (tid & 1) * 32;      /* 0 or 32 bf16 elems */             \
  const unsigned short* gA = (A_) + (size_t)(blockM + ldRow) * E_DIM + ldK;   \
  const unsigned short* gB = (W_) + (size_t)(blockN + ldRow) * E_DIM + ldK;   \
  unsigned ldsA[2], ldsB[2];   /* LDS byte addr = low 32b of flat ptr */      \
  ldsA[0] = (unsigned)(size_t)&sA[0][ldRow * TKL + ldK];                      \
  ldsA[1] = (unsigned)(size_t)&sA[1][ldRow * TKL + ldK];                      \
  ldsB[0] = (unsigned)(size_t)&sB[0][ldRow * TKL + ldK];                      \
  ldsB[1] = (unsigned)(size_t)&sB[1][ldRow * TKL + ldK];                      \
  v8f acc[4][2];                                                              \
  _Pragma("unroll") for (int mi = 0; mi < 4; ++mi)                            \
    _Pragma("unroll") for (int ni = 0; ni < 2; ++ni)                          \
      acc[mi][ni] = (v8f)0.0f;                                                \
  /* prologue: stream tile 0 into buffer 0 */                                 \
  ASYNC_COPY64B(ldsA[0], (unsigned long long)gA);                             \
  ASYNC_COPY64B(ldsB[0], (unsigned long long)gB);                             \
  for (int t = 0; t < NKT; ++t) {                                             \
    if (t + 1 < NKT) {                                                        \
      /* stream tile t+1 into the other buffer while computing on t */        \
      ASYNC_COPY64B(ldsA[(t + 1) & 1],                                        \
                    (unsigned long long)(gA + (t + 1) * TKL));                \
      ASYNC_COPY64B(ldsB[(t + 1) & 1],                                        \
                    (unsigned long long)(gB + (t + 1) * TKL));                \
      /* loads retire in order: <=8 outstanding => tile t has landed */       \
      asm volatile("s_wait_asynccnt 0x8" ::: "memory");                       \
    } else {                                                                  \
      asm volatile("s_wait_asynccnt 0x0" ::: "memory");                       \
    }                                                                         \
    __syncthreads();                                                          \
    const unsigned short* bufA = sA[t & 1];                                   \
    const unsigned short* bufB = sB[t & 1];                                   \
    _Pragma("unroll") for (int kk = 0; kk < TKL; kk += 32) {                  \
      v16bf aF[4], bF[2];                                                     \
      _Pragma("unroll") for (int mi = 0; mi < 4; ++mi) {                      \
        const v8u* p = reinterpret_cast<const v8u*>(                          \
            &bufA[(wm * 64 + mi * 16 + (lane & 15)) * TKL + kk +              \
                  (lane >> 4) * 16]);                                         \
        aF[mi] = __builtin_bit_cast(v16bf, *p);                               \
      }                                                                       \
      _Pragma("unroll") for (int ni = 0; ni < 2; ++ni) {                      \
        const v8u* p = reinterpret_cast<const v8u*>(                          \
            &bufB[(wn * 32 + ni * 16 + (lane & 15)) * TKL + kk +              \
                  (lane >> 4) * 16]);                                         \
        bF[ni] = __builtin_bit_cast(v16bf, *p);                               \
      }                                                                       \
      _Pragma("unroll") for (int mi = 0; mi < 4; ++mi)                        \
        _Pragma("unroll") for (int ni = 0; ni < 2; ++ni)                      \
          acc[mi][ni] = __builtin_amdgcn_wmma_f32_16x16x32_bf16(              \
              false, aF[mi], false, bF[ni], (short)0, acc[mi][ni],            \
              false, false);                                                  \
    }                                                                         \
    __syncthreads(); /* protect buffer t&1 before next issue */           \
  }                                                                           \
  const int colHalf = lane & 15;                                              \
  const int rowSel  = (lane >> 4) * 8;

// --- GEMM1: T = X @ Wv^T + bv, bf16 output ---------------------------------
__global__ __launch_bounds__(256) void gemm_bf16_bias_bf16out(
    const unsigned short* __restrict__ A, const unsigned short* __restrict__ W,
    const float* __restrict__ bias, unsigned short* __restrict__ C) {
  GEMM_MAINLOOP(A, W)
  #pragma unroll
  for (int ni = 0; ni < 2; ++ni) {
    int col  = blockN + wn * 32 + ni * 16 + colHalf;
    float bv = bias[col];
    #pragma unroll
    for (int mi = 0; mi < 4; ++mi) {
      int row0 = blockM + wm * 64 + mi * 16 + rowSel;
      #pragma unroll
      for (int r = 0; r < 8; ++r) {
        float v = acc[mi][ni][r] + bv;
        C[(size_t)(row0 + r) * E_DIM + col] = f32_to_bf16(v);
      }
    }
  }
}

// --- GEMM2: out = resid + scale*(T @ Wout^T + bout), f32 output -------------
__global__ __launch_bounds__(256) void gemm_bf16_bias_resid_f32out(
    const unsigned short* __restrict__ A, const unsigned short* __restrict__ W,
    const float* __restrict__ bias, const float* __restrict__ resid,
    const float* __restrict__ scalePtr, float* __restrict__ out) {
  GEMM_MAINLOOP(A, W)
  const float scale = *scalePtr;
  #pragma unroll
  for (int ni = 0; ni < 2; ++ni) {
    int col  = blockN + wn * 32 + ni * 16 + colHalf;
    float bv = bias[col];
    #pragma unroll
    for (int mi = 0; mi < 4; ++mi) {
      int row0 = blockM + wm * 64 + mi * 16 + rowSel;
      #pragma unroll
      for (int r = 0; r < 8; ++r) {
        size_t idx = (size_t)(row0 + r) * E_DIM + col;
        out[idx] = resid[idx] + scale * (acc[mi][ni][r] + bv);
      }
    }
  }
}

// ----------------------------- row LayerNorm --------------------------------
__global__ __launch_bounds__(256) void layernorm_inplace_kernel(
    float* __restrict__ y, const float* __restrict__ g,
    const float* __restrict__ b) {
  const int tid  = threadIdx.x;
  const int lane = tid & 31;
  const int wv   = tid >> 5;
  float* row = y + (size_t)blockIdx.x * E_DIM;

  float4 v = reinterpret_cast<float4*>(row)[tid];
  float s  = v.x + v.y + v.z + v.w;
  float sq = v.x * v.x + v.y * v.y + v.z * v.z + v.w * v.w;
  #pragma unroll
  for (int off = 16; off > 0; off >>= 1) {
    s  += __shfl_xor(s,  off, 32);
    sq += __shfl_xor(sq, off, 32);
  }
  __shared__ float wsum[8], wsq[8];
  if (lane == 0) { wsum[wv] = s; wsq[wv] = sq; }
  __syncthreads();
  float ts = 0.0f, tq = 0.0f;
  #pragma unroll
  for (int i = 0; i < 8; ++i) { ts += wsum[i]; tq += wsq[i]; }

  const float mean = ts * (1.0f / E_DIM);
  const float var  = tq * (1.0f / E_DIM) - mean * mean;
  const float inv  = rsqrtf(var + LN_EPS);

  float4 gv = reinterpret_cast<const float4*>(g)[tid];
  float4 bv = reinterpret_cast<const float4*>(b)[tid];
  v.x = (v.x - mean) * inv * gv.x + bv.x;
  v.y = (v.y - mean) * inv * gv.y + bv.y;
  v.z = (v.z - mean) * inv * gv.z + bv.z;
  v.w = (v.w - mean) * inv * gv.w + bv.w;
  reinterpret_cast<float4*>(row)[tid] = v;
}

// ---------------------------------------------------------------------------
extern "C" void kernel_launch(void* const* d_in, const int* in_sizes, int n_in,
                              void* d_out, int out_size, void* d_ws,
                              size_t ws_size, hipStream_t stream) {
  (void)in_sizes; (void)n_in; (void)out_size; (void)ws_size;

  const float* user_emb  = (const float*)d_in[0];
  const float* item_emb  = (const float*)d_in[1];
  const float* u2i_in_w  = (const float*)d_in[2];
  const float* u2i_in_b  = (const float*)d_in[3];
  const float* u2i_out_w = (const float*)d_in[4];
  const float* u2i_out_b = (const float*)d_in[5];
  const float* i2u_in_w  = (const float*)d_in[6];
  const float* i2u_in_b  = (const float*)d_in[7];
  const float* i2u_out_w = (const float*)d_in[8];
  const float* i2u_out_b = (const float*)d_in[9];
  const float* user_g    = (const float*)d_in[10];
  const float* user_b    = (const float*)d_in[11];
  const float* item_g    = (const float*)d_in[12];
  const float* item_b    = (const float*)d_in[13];
  const float* alpha     = (const float*)d_in[14];
  const float* beta      = (const float*)d_in[15];

  const size_t BE = (size_t)B_DIM * E_DIM;
  const size_t EE = (size_t)E_DIM * E_DIM;

  // Workspace layout (all bf16)
  unsigned short* ws  = (unsigned short*)d_ws;
  unsigned short* Xu  = ws;            // user_emb  bf16   [BE]
  unsigned short* Xi  = Xu  + BE;      // item_emb  bf16   [BE]
  unsigned short* Tu  = Xi  + BE;      // V-proj (u2i)     [BE]
  unsigned short* Ti  = Tu  + BE;      // V-proj (i2u)     [BE]
  unsigned short* WvU = Ti  + BE;      // u2i Wv slice     [EE]
  unsigned short* WvI = WvU + EE;      // i2u Wv slice     [EE]
  unsigned short* WoU = WvI + EE;      // u2i out_w        [EE]
  unsigned short* WoI = WoU + EE;      // i2u out_w        [EE]

  float* outU = (float*)d_out;         // user_evolved [BE]
  float* outI = outU + BE;             // item_evolved [BE]

  // 1) precision conversion to bf16 (weights: only the live Wv slice)
  const int nBE8 = (int)(BE / 8), nEE8 = (int)(EE / 8);
  pack_bf16_kernel<<<nBE8 / 256, 256, 0, stream>>>(user_emb, Xu, nBE8);
  pack_bf16_kernel<<<nBE8 / 256, 256, 0, stream>>>(item_emb, Xi, nBE8);
  pack_bf16_kernel<<<nEE8 / 256, 256, 0, stream>>>(u2i_in_w + 2 * EE, WvU, nEE8);
  pack_bf16_kernel<<<nEE8 / 256, 256, 0, stream>>>(i2u_in_w + 2 * EE, WvI, nEE8);
  pack_bf16_kernel<<<nEE8 / 256, 256, 0, stream>>>(u2i_out_w, WoU, nEE8);
  pack_bf16_kernel<<<nEE8 / 256, 256, 0, stream>>>(i2u_out_w, WoI, nEE8);

  // 2) V projections: Tu = item @ WvU^T + bv ; Ti = user @ WvI^T + bv
  dim3 gg(E_DIM / TN, B_DIM / TM);  // (8, 256)
  gemm_bf16_bias_bf16out<<<gg, 256, 0, stream>>>(Xi, WvU, u2i_in_b + 2 * E_DIM, Tu);
  gemm_bf16_bias_bf16out<<<gg, 256, 0, stream>>>(Xu, WvI, i2u_in_b + 2 * E_DIM, Ti);

  // 3) output projections fused with residual: out = x + scale*(T @ Wo^T + b)
  gemm_bf16_bias_resid_f32out<<<gg, 256, 0, stream>>>(Tu, WoU, u2i_out_b,
                                                      user_emb, alpha, outU);
  gemm_bf16_bias_resid_f32out<<<gg, 256, 0, stream>>>(Ti, WoI, i2u_out_b,
                                                      item_emb, beta, outI);

  // 4) row-wise LayerNorm, in place on d_out
  layernorm_inplace_kernel<<<B_DIM, 256, 0, stream>>>(outU, user_g, user_b);
  layernorm_inplace_kernel<<<B_DIM, 256, 0, stream>>>(outI, item_g, item_b);
}
